// IMPChat_75282186764832
// MI455X (gfx1250) — matrix operand order repaired and checked by reading, not compiled, not run on hardware
//
#include <hip/hip_runtime.h>
#include <math.h>

typedef unsigned short u16;
typedef __attribute__((ext_vector_type(16))) __bf16 v16bf;
typedef __attribute__((ext_vector_type(8)))  float  v8f;

#define BP 1280
#define LL 64
#define EMB 200
#define DP 224
#define KSTEPS (DP / 32)   // 7
#define LDT 66             // padded row stride for transposed V tile (bank-conflict avoidance)

// ---------- scalar bf16 helpers (RNE) ----------
__device__ __forceinline__ u16 f32_to_bf16(float x) {
    union { float f; unsigned int u; } v; v.f = x;
    unsigned int r = v.u + 0x7FFFu + ((v.u >> 16) & 1u);
    return (u16)(r >> 16);
}
__device__ __forceinline__ float bf16_to_f32(u16 h) {
    union { unsigned int u; float f; } v; v.u = ((unsigned int)h) << 16; return v.f;
}

union FragBF { v16bf v; unsigned int u[8]; u16 h[16]; };

// ---------- WMMA fragment loaders (CDNA5 wave32 layouts, CDNA5 ISA) ----------
// A: 16x32 bf16 tile, row-major source, lane m holds K pairs per documented split.
__device__ __forceinline__ v16bf load_frag_a(const u16* base, int ld, int row0, int k0, int lane) {
    const int half = (lane >> 4) & 1, m = lane & 15;
    const u16* row = base + (size_t)(row0 + m) * ld + k0;
    FragBF f;
#pragma unroll
    for (int v = 0; v < 8; ++v) {
        const int k = (v < 4) ? (2 * v + (half ? 8 : 0))
                              : (16 + 2 * (v - 4) + (half ? 8 : 0));
        f.u[v] = *(const unsigned int*)(row + k);   // two contiguous bf16 per VGPR
    }
    return f.v;
}

// B: 32x16 tile from a matrix stored [n][k] ("column-major for B"): contiguous K run per lane.
// lanes 0-15 -> K=k0+0..7,16..23; lanes 16-31 -> K=k0+8..15,24..31 (matches A fragment order).
__device__ __forceinline__ v16bf load_frag_b_cm(const u16* base, int ld, int k0, int n0, int lane) {
    const int half = (lane >> 4) & 1, n = lane & 15;
    const u16* p = base + (size_t)(n0 + n) * ld + k0;
    FragBF f;
#pragma unroll
    for (int v = 0; v < 8; ++v) f.u[v] = *(const unsigned int*)(p + ((v < 4) ? (2 * v + (half ? 8 : 0)) : (16 + 2 * (v - 4) + (half ? 8 : 0))));
    return f.v;
}

__device__ __forceinline__ v8f wmma_bf16(v16bf a, v16bf b, v8f c) {
    return __builtin_amdgcn_wmma_f32_16x16x32_bf16(false, a, false, b, (short)0, c, false, false);
}

// ---------- packing kernels ----------
__global__ void pack_act_kernel(const float* __restrict__ in, u16* __restrict__ out, long total) {
    long i = (long)blockIdx.x * blockDim.x + threadIdx.x;
    if (i >= total) return;
    int d = (int)(i % DP);
    long r = i / DP;
    out[i] = (d < EMB) ? f32_to_bf16(in[r * EMB + d]) : (u16)0;
}

// Pack a 200x200 fp32 weight into bf16 [n][k] (B-fragment layout), zero padded to 224x224.
// transposed=0: Wp[n][k] = in[n][k]   (for X @ W^T : B[k][n] = W[n][k])
// transposed=1: Wp[n][k] = in[k][n]   (for A @ W   : B[k][n] = W[k][n])
__global__ void pack_w_kernel(const float* __restrict__ in, u16* __restrict__ out, int transposed) {
    int i = blockIdx.x * blockDim.x + threadIdx.x;
    if (i >= DP * DP) return;
    int k = i % DP, n = i / DP;
    float v = 0.f;
    if (n < EMB && k < EMB) v = transposed ? in[k * EMB + n] : in[n * EMB + k];
    out[i] = f32_to_bf16(v);
}

// ---------- attention: O = Q + softmax(Q K^T * scale) V ----------
__global__ __launch_bounds__(256) void attn_kernel(
    const u16* __restrict__ Q, const u16* __restrict__ K, const u16* __restrict__ V,
    u16* __restrict__ O, float scale)
{
    __shared__ __align__(16) u16 sQ[LL * DP];
    __shared__ __align__(16) u16 sK[LL * DP];
    __shared__ __align__(16) u16 sVt[DP * LDT];   // V transposed: sVt[d][j]
    __shared__ __align__(16) float sS[LL * LL];
    __shared__ __align__(16) u16 sSb[LL * LL];

    const int b = blockIdx.x, tid = threadIdx.x;
    const size_t off = (size_t)b * LL * DP;
    const uint4* gq = (const uint4*)(Q + off);
    const uint4* gk = (const uint4*)(K + off);
    for (int i = tid; i < LL * DP / 8; i += 256) {
        ((uint4*)sQ)[i] = gq[i];
        ((uint4*)sK)[i] = gk[i];
    }
    // stage V transposed: coalesced global dword reads, conflict-free strided LDS writes
    {
        const unsigned int* gv = (const unsigned int*)(V + off);
        for (int i = tid; i < LL * DP / 2; i += 256) {
            const int row = (2 * i) / DP;
            const int d   = (2 * i) % DP;     // even
            const unsigned int x = gv[i];
            sVt[(d + 0) * LDT + row] = (u16)(x & 0xFFFFu);
            sVt[(d + 1) * LDT + row] = (u16)(x >> 16);
        }
    }
    __syncthreads();

    const int wave = tid >> 5, lane = tid & 31;
    const int half = (lane >> 4) & 1, cn = lane & 15;

    // S = scale * Q K^T  (16 tiles of 16x16; 2 per wave; K reads are contiguous row-major)
    for (int t = wave * 2; t < wave * 2 + 2; ++t) {
        const int mi = (t >> 2) << 4, ni = (t & 3) << 4;
        v8f acc = {};
        for (int kk = 0; kk < KSTEPS; ++kk)
            acc = wmma_bf16(load_frag_a(sQ, DP, mi, kk * 32, lane),
                            load_frag_b_cm(sK, DP, kk * 32, ni, lane), acc);
#pragma unroll
        for (int r = 0; r < 8; ++r)
            sS[(mi + r + 8 * half) * LL + ni + cn] = acc[r] * scale;
    }
    __syncthreads();

    // row softmax (one thread per row)
    if (tid < LL) {
        float* row = sS + tid * LL;
        float mx = row[0];
        for (int j = 1; j < LL; ++j) mx = fmaxf(mx, row[j]);
        float sum = 0.f;
        for (int j = 0; j < LL; ++j) { float e = __expf(row[j] - mx); row[j] = e; sum += e; }
        const float inv = 1.0f / sum;
        for (int j = 0; j < LL; ++j) sSb[tid * LL + j] = f32_to_bf16(row[j] * inv);
    }
    __syncthreads();

    // O = Q + S @ V   (4 x 14 = 56 tiles; 7 per wave; B from transposed V -> contiguous)
    u16* Ob = O + off;
    for (int t = wave * 7; t < wave * 7 + 7; ++t) {
        const int mi = (t / 14) << 4, ni = (t % 14) << 4;
        v8f acc = {};
#pragma unroll
        for (int r = 0; r < 8; ++r)
            acc[r] = bf16_to_f32(sQ[(mi + r + 8 * half) * DP + ni + cn]);
        for (int kk = 0; kk < 2; ++kk)
            acc = wmma_bf16(load_frag_a(sSb, LL, mi, kk * 32, lane),
                            load_frag_b_cm(sVt, LDT, kk * 32, ni, lane), acc);
#pragma unroll
        for (int r = 0; r < 8; ++r)
            Ob[(mi + r + 8 * half) * DP + ni + cn] = f32_to_bf16(acc[r]);
    }
}

// ---------- FFN: Out = relu(X W1^T + b1) W2^T + b2 + X ----------
__global__ __launch_bounds__(256) void ffn_kernel(
    const u16* __restrict__ X, const u16* __restrict__ W1, const float* __restrict__ b1,
    const u16* __restrict__ W2, const float* __restrict__ b2, u16* __restrict__ Out)
{
    __shared__ __align__(16) u16 sX[LL * DP];
    __shared__ __align__(16) u16 sH[LL * DP];
    const int b = blockIdx.x, tid = threadIdx.x;
    const size_t off = (size_t)b * LL * DP;
    const uint4* gx = (const uint4*)(X + off);
    for (int i = tid; i < LL * DP / 8; i += 256) ((uint4*)sX)[i] = gx[i];

    // warm the cache for the streamed weights (global_prefetch_b8 path)
    for (int i = tid; i < (DP * DP) / 64; i += 256) {
        __builtin_prefetch((const char*)(W1 + i * 64), 0, 3);
        __builtin_prefetch((const char*)(W2 + i * 64), 0, 3);
    }
    __syncthreads();

    const int wave = tid >> 5, lane = tid & 31;
    const int half = (lane >> 4) & 1, cn = lane & 15;

    for (int t = wave * 7; t < wave * 7 + 7; ++t) {
        const int mi = (t / 14) << 4, ni = (t % 14) << 4;
        v8f acc = {};
        for (int kk = 0; kk < KSTEPS; ++kk)
            acc = wmma_bf16(load_frag_a(sX, DP, mi, kk * 32, lane),
                            load_frag_b_cm(W1, DP, kk * 32, ni, lane), acc);
        const int ng = ni + cn;
        const float bias = (ng < EMB) ? b1[ng] : 0.f;
#pragma unroll
        for (int r = 0; r < 8; ++r)
            sH[(mi + r + 8 * half) * DP + ng] = f32_to_bf16(fmaxf(acc[r] + bias, 0.f));
    }
    __syncthreads();

    u16* Ob = Out + off;
    for (int t = wave * 7; t < wave * 7 + 7; ++t) {
        const int mi = (t / 14) << 4, ni = (t % 14) << 4;
        const int ng = ni + cn;
        const float bias = (ng < EMB) ? b2[ng] : 0.f;
        v8f acc = {};
#pragma unroll
        for (int r = 0; r < 8; ++r)
            acc[r] = bf16_to_f32(sX[(mi + r + 8 * half) * DP + ng]) + bias;
        for (int kk = 0; kk < KSTEPS; ++kk)
            acc = wmma_bf16(load_frag_a(sH, DP, mi, kk * 32, lane),
                            load_frag_b_cm(W2, DP, kk * 32, ni, lane), acc);
#pragma unroll
        for (int r = 0; r < 8; ++r)
            Ob[(mi + r + 8 * half) * DP + ng] = f32_to_bf16(acc[r]);
    }
}

// ---------- distance: M1 = (A Bm) C^T / sqrt(200); M2 = A C^T / (|A||C| + eps) ----------
__global__ __launch_bounds__(256) void distance_kernel(
    const u16* __restrict__ A, const u16* __restrict__ C, const u16* __restrict__ Bm,
    float* __restrict__ out, int ch, float inv_s200)
{
    __shared__ __align__(16) u16 sA[LL * DP];
    __shared__ __align__(16) u16 sC[LL * DP];
    __shared__ __align__(16) u16 sT[LL * DP];
    __shared__ float nA[LL];
    __shared__ float nC[LL];

    const int b = blockIdx.x, tid = threadIdx.x;
    const size_t off = (size_t)b * LL * DP;
    const uint4* ga = (const uint4*)(A + off);
    const uint4* gc = (const uint4*)(C + off);
    for (int i = tid; i < LL * DP / 8; i += 256) {
        ((uint4*)sA)[i] = ga[i];
        ((uint4*)sC)[i] = gc[i];
    }
    __syncthreads();

    const int wave = tid >> 5, lane = tid & 31;
    const int half = (lane >> 4) & 1, cn = lane & 15;

    // row norms (pad columns are exact zeros)
    if (tid < 2 * LL) {
        const int row = tid & (LL - 1);
        const u16* src = (tid < LL) ? sA : sC;
        float s = 0.f;
        for (int d = 0; d < DP; ++d) { const float v = bf16_to_f32(src[row * DP + d]); s += v * v; }
        if (tid < LL) nA[row] = sqrtf(s); else nC[row] = sqrtf(s);
    }

    // T = A @ Bm  (Bm pre-packed [n][k])
    for (int t = wave * 7; t < wave * 7 + 7; ++t) {
        const int mi = (t / 14) << 4, ni = (t % 14) << 4;
        v8f acc = {};
        for (int kk = 0; kk < KSTEPS; ++kk)
            acc = wmma_bf16(load_frag_a(sA, DP, mi, kk * 32, lane),
                            load_frag_b_cm(Bm, DP, kk * 32, ni, lane), acc);
#pragma unroll
        for (int r = 0; r < 8; ++r)
            sT[(mi + r + 8 * half) * DP + ni + cn] = f32_to_bf16(acc[r]);
    }
    __syncthreads();

    // M1/M2 fused: both need B fragments from C (row-major == [n][k] here)
    float* out1 = out + ((size_t)b * 8 + ch) * (LL * LL);
    float* out2 = out1 + LL * LL;
    for (int t = wave * 2; t < wave * 2 + 2; ++t) {
        const int mi = (t >> 2) << 4, ni = (t & 3) << 4;
        v8f a1 = {}, a2 = {};
        for (int kk = 0; kk < KSTEPS; ++kk) {
            const v16bf bfr = load_frag_b_cm(sC, DP, kk * 32, ni, lane);
            a1 = wmma_bf16(load_frag_a(sT, DP, mi, kk * 32, lane), bfr, a1);
            a2 = wmma_bf16(load_frag_a(sA, DP, mi, kk * 32, lane), bfr, a2);
        }
#pragma unroll
        for (int r = 0; r < 8; ++r) {
            const int m = mi + r + 8 * half, n = ni + cn;
            out1[m * LL + n] = a1[r] * inv_s200;
            out2[m * LL + n] = a2[r] / (nA[m] * nC[n] + 1e-6f);
        }
    }
}

// ---------- host orchestration ----------
extern "C" void kernel_launch(void* const* d_in, const int* in_sizes, int n_in,
                              void* d_out, int out_size, void* d_ws, size_t ws_size,
                              hipStream_t stream) {
    (void)in_sizes; (void)n_in; (void)out_size; (void)ws_size;

    const float* bU    = (const float*)d_in[0];
    const float* bR    = (const float*)d_in[1];
    const float* w_utt1= (const float*)d_in[2];
    const float* b_utt1= (const float*)d_in[3];
    const float* w_utt2= (const float*)d_in[4];
    const float* b_utt2= (const float*)d_in[5];
    const float* w_res1= (const float*)d_in[6];
    const float* b_res1= (const float*)d_in[7];
    const float* w_res2= (const float*)d_in[8];
    const float* b_res2= (const float*)d_in[9];
    const float* w_ur1 = (const float*)d_in[10];
    const float* b_ur1 = (const float*)d_in[11];
    const float* w_ur2 = (const float*)d_in[12];
    const float* b_ur2 = (const float*)d_in[13];
    const float* w_ru1 = (const float*)d_in[14];
    const float* b_ru1 = (const float*)d_in[15];
    const float* w_ru2 = (const float*)d_in[16];
    const float* b_ru2 = (const float*)d_in[17];
    const float* A1    = (const float*)d_in[18];
    const float* A2    = (const float*)d_in[19];
    float* out = (float*)d_out;

    char* ws = (char*)d_ws;
    const size_t actB = (size_t)BP * LL * DP * sizeof(u16);
    u16* Hu  = (u16*)(ws);
    u16* Hr  = (u16*)(ws + 1 * actB);
    u16* Hur = (u16*)(ws + 2 * actB);
    u16* Hru = (u16*)(ws + 3 * actB);
    u16* Tmp = (u16*)(ws + 4 * actB);
    u16* W   = (u16*)(ws + 5 * actB);
    const size_t wN = (size_t)DP * DP;
    u16* Wutt1 = W + 0 * wN; u16* Wutt2 = W + 1 * wN;
    u16* Wres1 = W + 2 * wN; u16* Wres2 = W + 3 * wN;
    u16* Wur1  = W + 4 * wN; u16* Wur2  = W + 5 * wN;
    u16* Wru1  = W + 6 * wN; u16* Wru2  = W + 7 * wN;
    u16* WA1   = W + 8 * wN; u16* WA2   = W + 9 * wN;

    const long tot = (long)BP * LL * DP;
    const unsigned anb = (unsigned)((tot + 255) / 256);
    pack_act_kernel<<<anb, 256, 0, stream>>>(bU, Hu, tot);
    pack_act_kernel<<<anb, 256, 0, stream>>>(bR, Hr, tot);

    const unsigned wnb = (DP * DP + 255) / 256;
    pack_w_kernel<<<wnb, 256, 0, stream>>>(w_utt1, Wutt1, 0);
    pack_w_kernel<<<wnb, 256, 0, stream>>>(w_utt2, Wutt2, 0);
    pack_w_kernel<<<wnb, 256, 0, stream>>>(w_res1, Wres1, 0);
    pack_w_kernel<<<wnb, 256, 0, stream>>>(w_res2, Wres2, 0);
    pack_w_kernel<<<wnb, 256, 0, stream>>>(w_ur1,  Wur1,  0);
    pack_w_kernel<<<wnb, 256, 0, stream>>>(w_ur2,  Wur2,  0);
    pack_w_kernel<<<wnb, 256, 0, stream>>>(w_ru1,  Wru1,  0);
    pack_w_kernel<<<wnb, 256, 0, stream>>>(w_ru2,  Wru2,  0);
    pack_w_kernel<<<wnb, 256, 0, stream>>>(A1, WA1, 1);
    pack_w_kernel<<<wnb, 256, 0, stream>>>(A2, WA2, 1);

    const float scale   = 1.0f / (sqrtf(200.0f) + 1e-8f);
    const float invs200 = 1.0f / sqrtf(200.0f);

    distance_kernel<<<BP, 256, 0, stream>>>(Hu, Hr, WA1, out, 0, invs200);

    for (int l = 0; l < 3; ++l) {
        attn_kernel<<<BP, 256, 0, stream>>>(Hu, Hu, Hu, Tmp, scale);
        ffn_kernel <<<BP, 256, 0, stream>>>(Tmp, Wutt1, b_utt1, Wutt2, b_utt2, Hu);
        attn_kernel<<<BP, 256, 0, stream>>>(Hr, Hr, Hr, Tmp, scale);
        ffn_kernel <<<BP, 256, 0, stream>>>(Tmp, Wres1, b_res1, Wres2, b_res2, Hr);
        attn_kernel<<<BP, 256, 0, stream>>>(Hu, Hr, Hr, Tmp, scale);
        ffn_kernel <<<BP, 256, 0, stream>>>(Tmp, Wur1, b_ur1, Wur2, b_ur2, Hur);
        attn_kernel<<<BP, 256, 0, stream>>>(Hr, Hu, Hu, Tmp, scale);
        ffn_kernel <<<BP, 256, 0, stream>>>(Tmp, Wru1, b_ru1, Wru2, b_ru2, Hru);
        distance_kernel<<<BP, 256, 0, stream>>>(Hur, Hru, WA2, out, 2 + 2 * l, invs200);
    }
}
